// RelationReasoning_74174085202627
// MI455X (gfx1250) — matrix operand order repaired and checked by reading, not compiled, not run on hardware
//
#include <hip/hip_runtime.h>
#include <hip/hip_bf16.h>

// ---------------------------------------------------------------------------
// MI455X (gfx1250): bf16 WMMA GEMMs + TDM-staged flash attention (wave32).
// ---------------------------------------------------------------------------

typedef __bf16 bf16_t;
typedef __attribute__((ext_vector_type(8)))  bf16_t v8bf;
typedef __attribute__((ext_vector_type(16))) bf16_t v16bf;
typedef __attribute__((ext_vector_type(8)))  float  v8f;
typedef __attribute__((ext_vector_type(4)))  unsigned int u32x4;
typedef __attribute__((ext_vector_type(8)))  int i32x8;
typedef __attribute__((ext_vector_type(4)))  int i32x4;

union ABf { v16bf v; v8bf h[2]; };

#define BATCH   8
#define LSEQ    1024
#define DMODEL  1024
#define HEADS   4
#define DHEAD   256
#define NCLS    1231      // NUM_CLASSES + 1
#define NCLS_PAD 1280     // padded to multiple of 64 for GEMM tiling

__device__ __forceinline__ v8f wmma_bf16(const ABf& a, const ABf& b, v8f c) {
  return __builtin_amdgcn_wmma_f32_16x16x32_bf16(
      false, a.v, false, b.v, (short)0, c, false, false);
}

// Tensor Data Mover: 2-D tile (tile_d0 contiguous elems x tile_d1 rows) of
// 2-byte elements from global (row pitch = row_stride_elems) into LDS at
// byte offset lds_off (rows packed contiguously). D# per CDNA5 ISA ch.8.
__device__ __forceinline__ void tdm_load_2d(unsigned lds_off, const void* gptr,
                                            unsigned tile_d0, unsigned tile_d1,
                                            unsigned row_stride_elems,
                                            unsigned tensor_d0) {
  unsigned long long ga = (unsigned long long)(size_t)gptr;
  u32x4 g0;
  g0[0] = 1u;                                  // count=1, user descriptor
  g0[1] = lds_off;                             // LDS byte address
  g0[2] = (unsigned)ga;                        // global_addr[31:0]
  g0[3] = (unsigned)((ga >> 32) & 0x01ffffffu) | (2u << 30);  // [56:32]|type=2
  i32x8 g1;
  g1[0] = (int)(1u << 16);                     // wg_mask=0, data_size=1 (2B)
  g1[1] = (int)((tensor_d0 & 0xffffu) << 16);  // tensor_dim0[15:0]
  g1[2] = (int)(tensor_d0 >> 16);              // tensor_dim0[31:16], dim1 lo=0
  g1[3] = (int)(16u | (tile_d0 << 16));        // tensor_dim1=1<<20, tile_dim0
  g1[4] = (int)(tile_d1 & 0xffffu);            // tile_dim1, tile_dim2=0
  g1[5] = (int)row_stride_elems;               // tensor_dim0_stride[31:0]
  g1[6] = 0;
  g1[7] = 0;
  i32x4 z4; z4[0] = 0; z4[1] = 0; z4[2] = 0; z4[3] = 0;
  i32x8 z8;
#pragma unroll
  for (int i = 0; i < 8; i++) z8[i] = 0;
  __builtin_amdgcn_tensor_load_to_lds(g0, g1, z4, z4, z8, 0);
}

// ---------------------------------------------------------------------------
// Packing / elementwise kernels
// ---------------------------------------------------------------------------

__global__ void cvt_f32_bf16_kernel(const float* __restrict__ src,
                                    bf16_t* __restrict__ dst, long n) {
  long i = (long)blockIdx.x * blockDim.x + threadIdx.x;
  if (i < n) dst[i] = (bf16_t)src[i];
}

__global__ void copy_f32_kernel(const float* __restrict__ src,
                                float* __restrict__ dst, long n) {
  long i = (long)blockIdx.x * blockDim.x + threadIdx.x;
  if (i < n) dst[i] = src[i];
}

__global__ void relu_cvt_kernel(const float* __restrict__ src,
                                bf16_t* __restrict__ dst, long n) {
  long i = (long)blockIdx.x * blockDim.x + threadIdx.x;
  if (i < n) dst[i] = (bf16_t)fmaxf(src[i], 0.0f);
}

// W[h]: [Kd, N] row-major f32 -> Wt[h]: [Np, Kd] row-major bf16 (transposed)
__global__ void pack_wt_kernel(const float* __restrict__ W, long wstride_h,
                               int Kd, int N, int Np,
                               bf16_t* __restrict__ Wt, long ostride_h) {
  long total = (long)Np * Kd;
  long i = (long)blockIdx.x * blockDim.x + threadIdx.x;
  if (i >= total) return;
  int h = blockIdx.z;
  long n = i / Kd;
  long k = i - n * Kd;
  float v = (n < N) ? W[(long)h * wstride_h + k * (long)N + n] : 0.0f;
  Wt[(long)h * ostride_h + i] = (bf16_t)v;
}

// ---------------------------------------------------------------------------
// bf16 GEMM: C[M,Npad] = A[M,Kd] * Bt[Npad,Kd]^T + bias
// 8 waves/block; wave computes 32x64 tile. Explicit 2-stage software pipeline
// with statically-named register buffers (Kd must be a multiple of 64).
// mode 0: bf16 out row-major (ld=Npad) + h*ostride_h
// mode 1: bf16 out "V-transpose": out[((h*BATCH+b)*Npad + n)*LSEQ + l]
// mode 2: f32 out row-major ld=Ncols, store only n < Ncols  (classifier)
// ---------------------------------------------------------------------------
__global__ __launch_bounds__(256, 1) void gemm_bf16_kernel(
    const bf16_t* __restrict__ A, int lda, long astride_h,
    const bf16_t* __restrict__ Bt, long bstride_h, int Kd,
    const float* __restrict__ bias, int bias_stride_h,
    void* __restrict__ outp, long ostride_h,
    int M, int Npad, int Ncols, int mode) {
  int h    = blockIdx.z;
  const bf16_t* Ah = A + (long)h * astride_h;
  const bf16_t* Bh = Bt + (long)h * bstride_h;
  const float*  bh = bias + (long)h * bias_stride_h;
  int wave = threadIdx.x >> 5;
  int lane = threadIdx.x & 31;
  int ln = lane & 15, lh = lane >> 4;
  int m0 = blockIdx.x * 256 + wave * 32;
  int n0 = blockIdx.y * 64;

  v8f acc[2][4];
#pragma unroll
  for (int nt = 0; nt < 4; nt++) {
    int nn = n0 + nt * 16 + ln;
    float bv = (nn < Ncols) ? bh[nn] : 0.0f;
#pragma unroll
    for (int mt = 0; mt < 2; mt++)
#pragma unroll
      for (int i = 0; i < 8; i++) acc[mt][nt][i] = bv;
  }

  const bf16_t* apr  = Ah + (long)(m0 + ln) * lda + 8 * lh;
  const bf16_t* apr2 = apr + 16 * lda;
  auto loadA = [&](ABf* dst, int k0) {
    dst[0].h[0] = *(const v8bf*)(apr + k0);
    dst[0].h[1] = *(const v8bf*)(apr + k0 + 16);
    dst[1].h[0] = *(const v8bf*)(apr2 + k0);
    dst[1].h[1] = *(const v8bf*)(apr2 + k0 + 16);
  };
  auto loadB = [&](ABf* dst, int k0) {
#pragma unroll
    for (int nt = 0; nt < 4; nt++) {
      const bf16_t* bp = Bh + (long)(n0 + nt * 16 + ln) * Kd + k0 + 8 * lh;
      dst[nt].h[0] = *(const v8bf*)bp;
      dst[nt].h[1] = *(const v8bf*)(bp + 16);
    }
  };

  // Two named pipeline stages -> every array index is a constant, so all
  // fragments stay in VGPRs (no alloca/LDS demotion).
  ABf a0[2], a1[2], b0[4], b1[4];
  loadA(a0, 0);
  loadB(b0, 0);
  for (int k = 0; k < Kd; k += 64) {
    loadA(a1, k + 32);
    loadB(b1, k + 32);
#pragma unroll
    for (int mt = 0; mt < 2; mt++)
#pragma unroll
      for (int nt = 0; nt < 4; nt++)
        acc[mt][nt] = wmma_bf16(a0[mt], b0[nt], acc[mt][nt]);
    if (k + 64 < Kd) {
      loadA(a0, k + 64);
      loadB(b0, k + 64);
    }
#pragma unroll
    for (int mt = 0; mt < 2; mt++)
#pragma unroll
      for (int nt = 0; nt < 4; nt++)
        acc[mt][nt] = wmma_bf16(a1[mt], b1[nt], acc[mt][nt]);
  }

#pragma unroll
  for (int mt = 0; mt < 2; mt++) {
#pragma unroll
    for (int nt = 0; nt < 4; nt++) {
#pragma unroll
      for (int r = 0; r < 8; r++) {
        int mm = m0 + mt * 16 + r + 8 * lh;
        int nn = n0 + nt * 16 + ln;
        float val = acc[mt][nt][r];
        if (mode == 0) {
          ((bf16_t*)outp)[(long)h * ostride_h + (long)mm * Npad + nn] =
              (bf16_t)val;
        } else if (mode == 1) {
          int b = mm >> 10;
          int l = mm & (LSEQ - 1);
          ((bf16_t*)outp)[(((long)h * BATCH + b) * (long)Npad + nn) * LSEQ + l] =
              (bf16_t)val;
        } else {
          if (nn < Ncols) ((float*)outp)[(long)mm * Ncols + nn] = val;
        }
      }
    }
  }
}

// ---------------------------------------------------------------------------
// Flash attention with TDM-staged K/V tiles.
// Block = 8 waves sharing (b, h, dslice); wave w handles q-tile qhi*8+w.
// Dynamic LDS map (base 0, no static shared):
//   [0,16K)  K tile buf0   [16K,32K) K tile buf1     (32 keys x 256 dh, bf16)
//   [32K,48K) V tile buf0  [48K,64K) V tile buf1     (256 d x 32 keys, bf16)
//   [64K,72K) per-wave P scratch (16x32 bf16 each)
// Wave 0 DMAs tile kt+1 while all waves compute on tile kt (TENSORcnt<=2).
// ---------------------------------------------------------------------------
__global__ __launch_bounds__(256, 1) void attention_kernel(
    const bf16_t* __restrict__ Qb, const bf16_t* __restrict__ Kb,
    const bf16_t* __restrict__ Vt, float* __restrict__ o) {
  extern __shared__ char smem[];
  int wave = threadIdx.x >> 5;
  int lane = threadIdx.x & 31;
  int ln = lane & 15, lh = lane >> 4;
  int grp    = blockIdx.x;
  int qhi    = grp & 7;
  int dslice = (grp >> 3) & 3;
  int h      = (grp >> 5) & 3;
  int b      = grp >> 7;
  int qt     = qhi * 8 + wave;

  const bf16_t* Qh = Qb + ((long)h * BATCH + b) * (long)LSEQ * DHEAD;
  const bf16_t* Kh = Kb + ((long)h * BATCH + b) * (long)LSEQ * DHEAD;
  const bf16_t* Vh = Vt + ((long)h * BATCH + b) * (long)DMODEL * LSEQ +
                     (long)(dslice * 256) * LSEQ;

  // Q A-fragments: 8 chunks of 16x32 covering DHEAD=256
  ABf qa[8];
  {
    const bf16_t* qp = Qh + (long)(qt * 16 + ln) * DHEAD + 8 * lh;
#pragma unroll
    for (int c = 0; c < 8; c++) {
      qa[c].h[0] = *(const v8bf*)(qp + c * 32);
      qa[c].h[1] = *(const v8bf*)(qp + c * 32 + 16);
    }
  }

  v8f acc[16];
#pragma unroll
  for (int nt = 0; nt < 16; nt++)
#pragma unroll
    for (int i = 0; i < 8; i++) acc[nt][i] = 0.0f;

  float mrow[8], lrow[8];
#pragma unroll
  for (int r = 0; r < 8; r++) { mrow[r] = -3.0e38f; lrow[r] = 0.0f; }

  const float cexp = 0.0901684401f;  // log2(e) / sqrt(DHEAD)
  bf16_t* pl = (bf16_t*)(smem + 65536) + wave * 512;

  // prologue DMA for kt = 0 into buffer 0
  if (wave == 0) {
    tdm_load_2d(0u, Kh, 256, 32, 256, 256);
    tdm_load_2d(32768u, Vh, 32, 256, 1024, 1024);
  }

  for (int kt = 0; kt < 32; kt++) {
    int cur = kt & 1;
    if (wave == 0) {
      if (kt + 1 < 32) {
        unsigned nb = (unsigned)((cur ^ 1) * 16384);
        tdm_load_2d(nb, Kh + (long)(kt + 1) * 32 * DHEAD, 256, 32, 256, 256);
        tdm_load_2d(32768u + nb, Vh + (kt + 1) * 32, 32, 256, 1024, 1024);
        __builtin_amdgcn_s_wait_tensorcnt(2);  // current pair complete
      } else {
        __builtin_amdgcn_s_wait_tensorcnt(0);
      }
    }
    __syncthreads();  // tile kt visible to all waves

    const bf16_t* Kl = (const bf16_t*)(smem + cur * 16384);
    const bf16_t* Vl = (const bf16_t*)(smem + 32768 + cur * 16384);

    v8f s0, s1;
#pragma unroll
    for (int i = 0; i < 8; i++) { s0[i] = 0.0f; s1[i] = 0.0f; }

    // scores: S[16q x 32k] = Q K^T (K rows from LDS tile)
#pragma unroll
    for (int c = 0; c < 8; c++) {
      ABf b0, b1;
      const bf16_t* kp0 = Kl + ln * 256 + c * 32 + 8 * lh;
      b0.h[0] = *(const v8bf*)kp0;
      b0.h[1] = *(const v8bf*)(kp0 + 16);
      const bf16_t* kp1 = kp0 + 16 * 256;
      b1.h[0] = *(const v8bf*)kp1;
      b1.h[1] = *(const v8bf*)(kp1 + 16);
      s0 = wmma_bf16(qa[c], b0, s0);
      s1 = wmma_bf16(qa[c], b1, s1);
    }

    // online softmax: each 16-lane half owns rows r + 8*lh
    float alpha[8];
#pragma unroll
    for (int r = 0; r < 8; r++) {
      float mx = fmaxf(s0[r], s1[r]);
#pragma unroll
      for (int msk = 1; msk < 16; msk <<= 1)
        mx = fmaxf(mx, __shfl_xor(mx, msk, 32));
      float mnew = fmaxf(mrow[r], mx);
      float al = exp2f((mrow[r] - mnew) * cexp);
      float p0 = exp2f((s0[r] - mnew) * cexp);
      float p1 = exp2f((s1[r] - mnew) * cexp);
      float rs = p0 + p1;
#pragma unroll
      for (int msk = 1; msk < 16; msk <<= 1) rs += __shfl_xor(rs, msk, 32);
      lrow[r] = lrow[r] * al + rs;
      mrow[r] = mnew;
      alpha[r] = al;
      int row = r + 8 * lh;
      pl[row * 32 + ln]      = (bf16_t)p0;  // C-layout -> row-major P in LDS
      pl[row * 32 + 16 + ln] = (bf16_t)p1;
    }
#pragma unroll
    for (int nt = 0; nt < 16; nt++)
#pragma unroll
      for (int r = 0; r < 8; r++) acc[nt][r] *= alpha[r];

    asm volatile("s_wait_dscnt 0x0" ::: "memory");  // per-wave P RAW

    ABf pa;  // reload P as 16x32 A-fragment
    const bf16_t* pp = pl + ln * 32 + 8 * lh;
    pa.h[0] = *(const v8bf*)pp;
    pa.h[1] = *(const v8bf*)(pp + 16);

    // O += P @ V (V rows = d, contiguous keys, from LDS tile)
#pragma unroll
    for (int nt = 0; nt < 16; nt++) {
      ABf vb;
      const bf16_t* vp = Vl + (nt * 16 + ln) * 32 + 8 * lh;
      vb.h[0] = *(const v8bf*)vp;
      vb.h[1] = *(const v8bf*)(vp + 16);
      acc[nt] = wmma_bf16(pa, vb, acc[nt]);
    }
    __syncthreads();  // all waves done with tile kt before its buffer is reused
  }

  float inv[8];
#pragma unroll
  for (int r = 0; r < 8; r++) inv[r] = 1.0f / lrow[r];
#pragma unroll
  for (int nt = 0; nt < 16; nt++) {
#pragma unroll
    for (int r = 0; r < 8; r++) {
      int row = qt * 16 + r + 8 * lh;
      int col = dslice * 256 + nt * 16 + ln;
      atomicAdd(&o[((long)b * LSEQ + row) * DMODEL + col], acc[nt][r] * inv[r]);
    }
  }
}

// ---------------------------------------------------------------------------
// Host-side orchestration
// ---------------------------------------------------------------------------
extern "C" void kernel_launch(void* const* d_in, const int* in_sizes, int n_in,
                              void* d_out, int out_size, void* d_ws,
                              size_t ws_size, hipStream_t stream) {
  (void)in_sizes; (void)n_in; (void)out_size; (void)ws_size;
  const float* x  = (const float*)d_in[0];
  const float* WQ = (const float*)d_in[1];
  const float* bQ = (const float*)d_in[2];
  const float* WK = (const float*)d_in[3];
  const float* bK = (const float*)d_in[4];
  const float* WV = (const float*)d_in[5];
  const float* bV = (const float*)d_in[6];
  const float* WO = (const float*)d_in[7];
  const float* bO = (const float*)d_in[8];
  const float* Wc = (const float*)d_in[9];
  const float* bc = (const float*)d_in[10];
  float* out = (float*)d_out;

  char* ws = (char*)d_ws;
  size_t off = 0;
  auto alloc = [&](size_t bytes) -> char* {
    char* p = ws + off;
    off += (bytes + 255) & ~(size_t)255;
    return p;
  };

  const long ML = (long)BATCH * LSEQ;  // 8192
  bf16_t* xb  = (bf16_t*)alloc(ML * DMODEL * 2);  // reused later for relu(o)
  bf16_t* WQt = (bf16_t*)alloc((long)HEADS * DHEAD * DMODEL * 2);
  bf16_t* WKt = (bf16_t*)alloc((long)HEADS * DHEAD * DMODEL * 2);
  bf16_t* WVt = (bf16_t*)alloc((long)HEADS * DMODEL * DHEAD * 2);
  bf16_t* WOt = (bf16_t*)alloc((long)DHEAD * DMODEL * 2);
  bf16_t* Wct = (bf16_t*)alloc((long)NCLS_PAD * DMODEL * 2);
  bf16_t* Oxb = (bf16_t*)alloc(ML * DHEAD * 2);
  bf16_t* Qb  = (bf16_t*)alloc((long)HEADS * ML * DHEAD * 2);
  bf16_t* Kb  = (bf16_t*)alloc((long)HEADS * ML * DHEAD * 2);
  bf16_t* Vt  = (bf16_t*)alloc((long)HEADS * BATCH * (long)DMODEL * LSEQ * 2);
  float*  oacc = (float*)alloc(ML * DMODEL * 4);

  long nX = ML * DMODEL;
  cvt_f32_bf16_kernel<<<dim3((nX + 255) / 256), dim3(256), 0, stream>>>(x, xb,
                                                                        nX);
  {
    long nqt = (long)DHEAD * DMODEL;
    dim3 g((unsigned)((nqt + 255) / 256), 1, HEADS);
    pack_wt_kernel<<<g, 256, 0, stream>>>(WQ, (long)DMODEL * DHEAD, DMODEL,
                                          DHEAD, DHEAD, WQt, nqt);
    pack_wt_kernel<<<g, 256, 0, stream>>>(WK, (long)DMODEL * DHEAD, DMODEL,
                                          DHEAD, DHEAD, WKt, nqt);
    long nvt = (long)DMODEL * DHEAD;
    dim3 gv((unsigned)((nvt + 255) / 256), 1, HEADS);
    pack_wt_kernel<<<gv, 256, 0, stream>>>(WV, (long)DHEAD * DMODEL, DHEAD,
                                           DMODEL, DMODEL, WVt, nvt);
    dim3 go((unsigned)((nqt + 255) / 256), 1, 1);
    pack_wt_kernel<<<go, 256, 0, stream>>>(WO, 0, DMODEL, DHEAD, DHEAD, WOt, 0);
    long nct = (long)NCLS_PAD * DMODEL;
    dim3 gc((unsigned)((nct + 255) / 256), 1, 1);
    pack_wt_kernel<<<gc, 256, 0, stream>>>(Wc, 0, DMODEL, NCLS, NCLS_PAD, Wct,
                                           0);
  }

  // Ox = x @ WO + bO
  gemm_bf16_kernel<<<dim3(ML / 256, DHEAD / 64, 1), 256, 0, stream>>>(
      xb, DMODEL, 0, WOt, 0, DMODEL, bO, 0, Oxb, 0, (int)ML, DHEAD, DHEAD, 0);
  // Q[h], K[h]
  gemm_bf16_kernel<<<dim3(ML / 256, DHEAD / 64, HEADS), 256, 0, stream>>>(
      xb, DMODEL, 0, WQt, (long)DHEAD * DMODEL, DMODEL, bQ, DHEAD, Qb,
      ML * DHEAD, (int)ML, DHEAD, DHEAD, 0);
  gemm_bf16_kernel<<<dim3(ML / 256, DHEAD / 64, HEADS), 256, 0, stream>>>(
      xb, DMODEL, 0, WKt, (long)DHEAD * DMODEL, DMODEL, bK, DHEAD, Kb,
      ML * DHEAD, (int)ML, DHEAD, DHEAD, 0);
  // V[h] stored transposed as Vt[h][b][d][l]
  gemm_bf16_kernel<<<dim3(ML / 256, DMODEL / 64, HEADS), 256, 0, stream>>>(
      Oxb, DHEAD, 0, WVt, (long)DMODEL * DHEAD, DHEAD, bV, DMODEL, Vt, 0,
      (int)ML, DMODEL, DMODEL, 1);

  // residual seed, attention, relu+cvt, classifier
  copy_f32_kernel<<<dim3((nX + 255) / 256), 256, 0, stream>>>(x, oacc, nX);
  attention_kernel<<<dim3(1024), 256, 73728, stream>>>(Qb, Kb, Vt, oacc);
  relu_cvt_kernel<<<dim3((nX + 255) / 256), 256, 0, stream>>>(oacc, xb, nX);
  gemm_bf16_kernel<<<dim3(ML / 256, NCLS_PAD / 64, 1), 256, 0, stream>>>(
      xb, DMODEL, 0, Wct, 0, DMODEL, bc, 0, out, 0, (int)ML, NCLS_PAD, NCLS, 2);
}
